// MultiHeadAttention_85031762526882
// MI455X (gfx1250) — matrix operand order repaired and checked by reading, not compiled, not run on hardware
//
#include <hip/hip_runtime.h>
#include <hip/hip_bf16.h>

typedef __bf16 bf16;
typedef __attribute__((ext_vector_type(8)))  float v8f;
typedef __attribute__((ext_vector_type(16))) bf16  v16bf;
typedef __attribute__((ext_vector_type(8)))  bf16  v8bf;
typedef __attribute__((ext_vector_type(4)))  bf16  v4bf;

static constexpr int Bb = 2, S = 2048, E = 2048, H = 16, Dh = 128;

__device__ __forceinline__ v16bf cat8(v8bf lo, v8bf hi) {
  return __builtin_shufflevector(lo, hi, 0,1,2,3,4,5,6,7,8,9,10,11,12,13,14,15);
}
#define WMMA_BF16(a, b, c) \
  __builtin_amdgcn_wmma_f32_16x16x32_bf16(false, (a), false, (b), (short)0, (c), false, false)

// ---------------------------------------------------------------- fp32 -> bf16
__global__ void cvt_f32_bf16(const float* __restrict__ src, bf16* __restrict__ dst, int n) {
  int i = (blockIdx.x * blockDim.x + threadIdx.x) * 4;
  if (i + 3 < n) {
    float4 v = *(const float4*)(src + i);
    v4bf o;
    o[0] = (bf16)v.x; o[1] = (bf16)v.y; o[2] = (bf16)v.z; o[3] = (bf16)v.w;
    *(v4bf*)(dst + i) = o;
  }
}

// ------------------------------------------------- C(MxN) = A(MxK) * W(NxK)^T
// bf16 inputs, fp32 output.  128x128 block tile, 256 threads = 8 waves,
// wave tile 32(M) x 64(N) = 2x4 WMMA accumulators, K step 32.
// Ping-pong LDS double buffering: ONE barrier per K step; global loads for
// tile k+1 are in flight while WMMAs consume tile k.
__global__ __launch_bounds__(256)
void gemm_bf16_nt(const bf16* __restrict__ A, const bf16* __restrict__ W,
                  float* __restrict__ C, int M, int N, int K) {
  __shared__ __align__(16) bf16 ldsA[2][128 * 32];
  __shared__ __align__(16) bf16 ldsB[2][128 * 32];
  const int tid  = threadIdx.x;
  const int lane = tid & 31;
  const int wave = tid >> 5;
  const int ln   = lane & 15;
  const int hi   = (lane >> 4) & 1;
  const int bm   = blockIdx.y * 128;
  const int bn   = blockIdx.x * 128;
  const int wm   = (wave >> 1) * 32;  // 0,32,64,96
  const int wn   = (wave & 1) * 64;   // 0,64

  v8f acc[2][4];
#pragma unroll
  for (int i = 0; i < 2; i++)
#pragma unroll
    for (int j = 0; j < 4; j++) acc[i][j] = (v8f){0.f,0.f,0.f,0.f,0.f,0.f,0.f,0.f};

  const int srow = tid >> 1;        // 0..127
  const int sseg = (tid & 1) * 16;  // 0 or 16
  const int sidx = srow * 32 + sseg;
  const bf16* gA = A + (size_t)(bm + srow) * K + sseg;
  const bf16* gB = W + (size_t)(bn + srow) * K + sseg;

  // prologue: stage tile k0=0 into buffer 0
  *(v8bf*)(&ldsA[0][sidx])     = *(const v8bf*)(gA);
  *(v8bf*)(&ldsA[0][sidx + 8]) = *(const v8bf*)(gA + 8);
  *(v8bf*)(&ldsB[0][sidx])     = *(const v8bf*)(gB);
  *(v8bf*)(&ldsB[0][sidx + 8]) = *(const v8bf*)(gB + 8);
  __syncthreads();

  int p = 0;
  for (int k0 = 0; k0 < K; k0 += 32) {
    // L2 prefetch two tiles ahead (global_prefetch_b8)
    __builtin_prefetch((const void*)(gA + k0 + 64), 0, 0);
    __builtin_prefetch((const void*)(gB + k0 + 64), 0, 0);
    // issue global loads for tile k0+32 (stays inside workspace on last iter)
    v8bf ra0 = *(const v8bf*)(gA + k0 + 32);
    v8bf ra1 = *(const v8bf*)(gA + k0 + 40);
    v8bf rb0 = *(const v8bf*)(gB + k0 + 32);
    v8bf rb1 = *(const v8bf*)(gB + k0 + 40);

    // compute on buffer p
    v16bf afrag[2];
#pragma unroll
    for (int mt = 0; mt < 2; mt++) {
      // A 16x32 layout: lanes<16 hold K0-15, lanes>=16 hold K16-31
      const bf16* rp = &ldsA[p][(wm + mt * 16 + ln) * 32];
      v8bf lo = *(const v8bf*)(rp + (hi ? 16 : 0));
      v8bf hb = *(const v8bf*)(rp + 8 + (hi ? 16 : 0));
      afrag[mt] = cat8(lo, hb);
    }
#pragma unroll
    for (int nt = 0; nt < 4; nt++) {
      // B 32x16 layout: lane n holds column n; lanes<16 K0-15, lanes>=16 K16-31
      const bf16* rp = &ldsB[p][(wn + nt * 16 + ln) * 32 + (hi ? 16 : 0)];
      v16bf bfrag = cat8(*(const v8bf*)rp, *(const v8bf*)(rp + 8));
      acc[0][nt] = WMMA_BF16(afrag[0], bfrag, acc[0][nt]);
      acc[1][nt] = WMMA_BF16(afrag[1], bfrag, acc[1][nt]);
    }

    // stage tile k0+32 into the other buffer (dead stores on final iter)
    const int q = p ^ 1;
    *(v8bf*)(&ldsA[q][sidx])     = ra0;
    *(v8bf*)(&ldsA[q][sidx + 8]) = ra1;
    *(v8bf*)(&ldsB[q][sidx])     = rb0;
    *(v8bf*)(&ldsB[q][sidx + 8]) = rb1;
    __syncthreads();
    p = q;
  }

#pragma unroll
  for (int mt = 0; mt < 2; mt++)
#pragma unroll
    for (int nt = 0; nt < 4; nt++)
#pragma unroll
      for (int j = 0; j < 8; j++) {
        int row = bm + wm + mt * 16 + j + 8 * hi;
        int col = bn + wn + nt * 16 + ln;
        C[(size_t)row * N + col] = acc[mt][nt][j];
      }
}

// ---------------------------------------------------------- RoPE on Q,K (fp32 -> bf16)
__global__ void rope_qk(const float* __restrict__ Qf, const float* __restrict__ Kf,
                        const float* __restrict__ cosp, const float* __restrict__ sinp,
                        bf16* __restrict__ Qo, bf16* __restrict__ Ko) {
  int idx = blockIdx.x * blockDim.x + threadIdx.x;  // B*S*H*64 = 2^22
  int d = idx & 63;
  int h = (idx >> 6) & (H - 1);
  int s = (idx >> 10) & (S - 1);
  int b = idx >> 21;
  size_t base = ((size_t)(b * S + s) * H + h) * Dh;
  float c = cosp[s * 64 + d], sn = sinp[s * 64 + d];
  float q1 = Qf[base + d], q2 = Qf[base + 64 + d];
  Qo[base + d]      = (bf16)(q1 * c - q2 * sn);
  Qo[base + 64 + d] = (bf16)(q2 * c + q1 * sn);
  float k1 = Kf[base + d], k2 = Kf[base + 64 + d];
  Ko[base + d]      = (bf16)(k1 * c - k2 * sn);
  Ko[base + 64 + d] = (bf16)(k2 * c + k1 * sn);
}

// ----------------------------------------------------------------- flash attention
// grid (S/64, B*H), 128 threads = 4 waves; each wave owns 16 q-rows.
__global__ __launch_bounds__(128)
void flash_attn(const bf16* __restrict__ Qp, const bf16* __restrict__ Kp,
                const bf16* __restrict__ Vp, float* __restrict__ Op) {
  __shared__ __align__(16) bf16 ldsK[32 * 128];   // [key][d]
  __shared__ __align__(16) bf16 ldsVT[128 * 32];  // [d][key]
  __shared__ __align__(16) bf16 ldsP[4][16 * 32]; // per-wave P scratch
  const int tid  = threadIdx.x;
  const int lane = tid & 31;
  const int wave = tid >> 5;
  const int ln   = lane & 15;
  const int hi   = (lane >> 4) & 1;
  const int b    = blockIdx.y >> 4;
  const int h    = blockIdx.y & 15;
  const int q0   = blockIdx.x * 64 + wave * 16;
  const float scaling = 0.08838834764831845f;  // 128^-0.5

  v16bf qfrag[4];
  {
    size_t qrow = ((size_t)(b * S + q0 + ln) * H + h) * Dh;
#pragma unroll
    for (int f = 0; f < 4; f++) {
      const bf16* p = Qp + qrow + f * 32 + (hi ? 16 : 0);
      qfrag[f] = cat8(*(const v8bf*)p, *(const v8bf*)(p + 8));
    }
  }

  v8f oacc[8];
#pragma unroll
  for (int t = 0; t < 8; t++) oacc[t] = (v8f){0.f,0.f,0.f,0.f,0.f,0.f,0.f,0.f};
  float mrow[8], lrow[8];
#pragma unroll
  for (int j = 0; j < 8; j++) { mrow[j] = -1e30f; lrow[j] = 0.f; }

  const int sr = tid >> 2;        // key row 0..31
  const int sc = (tid & 3) * 32;  // d segment

  for (int kv0 = 0; kv0 < S; kv0 += 32) {
    __syncthreads();
    {  // stage K tile and V^T tile
      size_t grow = ((size_t)(b * S + kv0 + sr) * H + h) * Dh + sc;
      if (kv0 + 32 < S) {  // prefetch next kv tile into L2
        __builtin_prefetch((const void*)(Kp + grow + (size_t)32 * H * Dh), 0, 0);
        __builtin_prefetch((const void*)(Vp + grow + (size_t)32 * H * Dh), 0, 0);
      }
#pragma unroll
      for (int c = 0; c < 32; c += 8)
        *(v8bf*)(ldsK + sr * 128 + sc + c) = *(const v8bf*)(Kp + grow + c);
#pragma unroll
      for (int c = 0; c < 32; c += 8) {
        v8bf vv = *(const v8bf*)(Vp + grow + c);
#pragma unroll
        for (int i = 0; i < 8; i++) ldsVT[(sc + c + i) * 32 + sr] = vv[i];
      }
    }
    __syncthreads();

    // scores: S0 = keys 0-15, S1 = keys 16-31 of this kv tile
    v8f s0 = (v8f){0.f,0.f,0.f,0.f,0.f,0.f,0.f,0.f};
    v8f s1 = (v8f){0.f,0.f,0.f,0.f,0.f,0.f,0.f,0.f};
#pragma unroll
    for (int f = 0; f < 4; f++) {
      const bf16* rp0 = ldsK + ln * 128 + f * 32 + (hi ? 16 : 0);
      v16bf kb0 = cat8(*(const v8bf*)rp0, *(const v8bf*)(rp0 + 8));
      s0 = WMMA_BF16(qfrag[f], kb0, s0);
      const bf16* rp1 = ldsK + (16 + ln) * 128 + f * 32 + (hi ? 16 : 0);
      v16bf kb1 = cat8(*(const v8bf*)rp1, *(const v8bf*)(rp1 + 8));
      s1 = WMMA_BF16(qfrag[f], kb1, s1);
    }

    // online softmax (row = one VGPR index across a 16-lane group)
    float p0[8], p1[8], alpha[8];
#pragma unroll
    for (int j = 0; j < 8; j++) {
      float a = s0[j] * scaling, c = s1[j] * scaling;
      float mj = fmaxf(a, c);
#pragma unroll
      for (int off = 1; off < 16; off <<= 1) mj = fmaxf(mj, __shfl_xor(mj, off, 32));
      float mnew = fmaxf(mrow[j], mj);
      alpha[j] = __expf(mrow[j] - mnew);
      p0[j] = __expf(a - mnew);
      p1[j] = __expf(c - mnew);
      float r = p0[j] + p1[j];
#pragma unroll
      for (int off = 1; off < 16; off <<= 1) r += __shfl_xor(r, off, 32);
      lrow[j] = lrow[j] * alpha[j] + r;
      mrow[j] = mnew;
    }
#pragma unroll
    for (int t = 0; t < 8; t++)
#pragma unroll
      for (int j = 0; j < 8; j++) oacc[t][j] *= alpha[j];

    // C-layout -> A-layout via per-wave LDS round trip
    bf16* pw = ldsP[wave];
#pragma unroll
    for (int j = 0; j < 8; j++) {
      int m = j + 8 * hi;
      pw[m * 32 + ln]      = (bf16)p0[j];
      pw[m * 32 + 16 + ln] = (bf16)p1[j];
    }
    asm volatile("s_wait_dscnt 0" ::: "memory");
    v16bf pfrag;
    {
      const bf16* rp = pw + ln * 32;
      pfrag = cat8(*(const v8bf*)(rp + (hi ? 16 : 0)),
                   *(const v8bf*)(rp + 8 + (hi ? 16 : 0)));
    }
    // O += P * V  (8 d-tiles of 16)
#pragma unroll
    for (int t = 0; t < 8; t++) {
      const bf16* rp = ldsVT + (t * 16 + ln) * 32 + (hi ? 16 : 0);
      v16bf vfrag = cat8(*(const v8bf*)rp, *(const v8bf*)(rp + 8));
      oacc[t] = WMMA_BF16(pfrag, vfrag, oacc[t]);
    }
  }

#pragma unroll
  for (int j = 0; j < 8; j++) {
    float inv = 1.f / lrow[j];
    int m = j + 8 * hi;
    size_t orow = ((size_t)(b * S + q0 + m) * H + h) * Dh;
#pragma unroll
    for (int t = 0; t < 8; t++) Op[orow + t * 16 + ln] = oacc[t][j] * inv;
  }
}

// ------------------------------------------------------------------------ launch
extern "C" void kernel_launch(void* const* d_in, const int* in_sizes, int n_in,
                              void* d_out, int out_size, void* d_ws, size_t ws_size,
                              hipStream_t stream) {
  (void)in_sizes; (void)n_in; (void)out_size; (void)ws_size;
  const float* hs   = (const float*)d_in[0];
  const float* cosp = (const float*)d_in[1];
  const float* sinp = (const float*)d_in[2];
  const float* Wq   = (const float*)d_in[3];
  const float* Wk   = (const float*)d_in[4];
  const float* Wv   = (const float*)d_in[5];
  const float* Wo   = (const float*)d_in[6];
  float* out = (float*)d_out;
  char*  ws  = (char*)d_ws;

  const int M = Bb * S;          // 4096
  const int BSE = Bb * S * E;    // 8388608
  const int EE  = E * E;         // 4194304

  bf16* Xb  = (bf16*)(ws + 0);
  bf16* Wqb = (bf16*)(ws + 16777216);
  bf16* Wkb = (bf16*)(ws + 25165824);
  bf16* Wvb = (bf16*)(ws + 33554432);
  bf16* Wob = (bf16*)(ws + 41943040);
  float* Qf = (float*)(ws + 50331648);
  float* Kf = (float*)(ws + 83886080);
  float* Vf = (float*)(ws + 117440512);
  bf16* Qb2 = (bf16*)(ws + 150994944);
  bf16* Kb2 = (bf16*)(ws + 167772160);
  bf16* Vb2 = (bf16*)(ws + 184549376);
  float* Of = Qf;                 // alias: Qf dead after rope
  bf16*  Ob = (bf16*)Kf;          // alias: Kf dead after rope

  cvt_f32_bf16<<<BSE / 1024, 256, 0, stream>>>(hs, Xb, BSE);
  cvt_f32_bf16<<<EE / 1024, 256, 0, stream>>>(Wq, Wqb, EE);
  cvt_f32_bf16<<<EE / 1024, 256, 0, stream>>>(Wk, Wkb, EE);
  cvt_f32_bf16<<<EE / 1024, 256, 0, stream>>>(Wv, Wvb, EE);
  cvt_f32_bf16<<<EE / 1024, 256, 0, stream>>>(Wo, Wob, EE);

  dim3 ggrid(E / 128, M / 128);
  gemm_bf16_nt<<<ggrid, 256, 0, stream>>>(Xb, Wqb, Qf, M, E, E);
  gemm_bf16_nt<<<ggrid, 256, 0, stream>>>(Xb, Wkb, Kf, M, E, E);
  gemm_bf16_nt<<<ggrid, 256, 0, stream>>>(Xb, Wvb, Vf, M, E, E);

  rope_qk<<<(Bb * S * H * 64) / 256, 256, 0, stream>>>(Qf, Kf, cosp, sinp, Qb2, Kb2);
  cvt_f32_bf16<<<BSE / 1024, 256, 0, stream>>>(Vf, Vb2, BSE);

  flash_attn<<<dim3(S / 64, Bb * H), 128, 0, stream>>>(Qb2, Kb2, Vb2, Of);

  cvt_f32_bf16<<<BSE / 1024, 256, 0, stream>>>(Of, Ob, BSE);
  gemm_bf16_nt<<<ggrid, 256, 0, stream>>>(Ob, Wob, out, M, E, E);
}
